// RWKV_TmixWrapper_26560077758560
// MI455X (gfx1250) — compile-verified, matrix operand order checked by reading, not run on hardware
//
#include <hip/hip_runtime.h>
#include <hip/hip_bf16.h>

#define B_  4
#define T_  1024
#define C_  1024
#define H_  16
#define HEAD_ 64
#define M_  (B_ * T_)      // 4096 rows
#define LORA_MIX 32
#define LORA_DECAY 64

typedef __attribute__((ext_vector_type(16))) __bf16 v16bf;
typedef __attribute__((ext_vector_type(8)))  float  v8f;
typedef __attribute__((ext_vector_type(4)))  unsigned v4u;
typedef __attribute__((ext_vector_type(8)))  int   v8i;
typedef __attribute__((ext_vector_type(4)))  int   v4i;

#if defined(__has_builtin)
#if __has_builtin(__builtin_amdgcn_tensor_load_to_lds) && __has_builtin(__builtin_amdgcn_s_wait_tensorcnt)
#define USE_TDM 1
#endif
#endif
#ifndef USE_TDM
#define USE_TDM 0
#endif

__device__ __forceinline__ unsigned short f2bf(float f) {
    unsigned u = __builtin_bit_cast(unsigned, f);
    u += 0x7FFFu + ((u >> 16) & 1u);           // round-to-nearest-even
    return (unsigned short)(u >> 16);
}

union FragBF { v16bf v; float4 f4[2]; };

// ---------------------------------------------------------------------------
// Transpose + convert fp32 [K x N] row-major -> bf16 [N x K] row-major
// ---------------------------------------------------------------------------
__global__ void convT_kernel(const float* __restrict__ W,
                             unsigned short* __restrict__ Wt,
                             int K, int N) {
    int idx = blockIdx.x * blockDim.x + threadIdx.x;
    if (idx >= K * N) return;
    int k = idx / N, n = idx % N;
    Wt[(size_t)n * K + k] = f2bf(W[idx]);
}

// ---------------------------------------------------------------------------
// Token shift: xx = shift(x)-x (fp32), xxx = bf16(x + xx*maa_x)
// ---------------------------------------------------------------------------
__global__ void prep_kernel(const float* __restrict__ x,
                            const float* __restrict__ maa_x,
                            float* __restrict__ xx,
                            unsigned short* __restrict__ xxx) {
    int idx = blockIdx.x * blockDim.x + threadIdx.x;
    if (idx >= M_ * C_) return;
    int c   = idx % C_;
    int row = idx / C_;
    int t   = row % T_;
    float xv = x[idx];
    float xp = (t == 0) ? 0.f : x[idx - C_];
    float xxv = xp - xv;
    xx[idx]  = xxv;
    xxx[idx] = f2bf(xv + xxv * maa_x[c]);
}

// ---------------------------------------------------------------------------
// bf16 GEMM with f32 accumulate: C[M,N] = A[M,K] * Bt[N,K]^T
// block = 256 threads (8 waves); block tile 128x64; wave tile 16x64.
// B tile (64 cols x 32 k, bf16) staged in LDS by the Tensor Data Mover,
// double-buffered; each wave DMAs its own 8-column slice and tracks it with
// TENSORcnt. A fragments come straight from global (b128, per-wave rows).
// ---------------------------------------------------------------------------
#define BTILE_ELEMS (64 * 32)          // 4 KB per buffer (bf16)

__device__ __forceinline__ void tdm_issue_bslice(const unsigned short* Bt, int ldb,
                                                 int Brows, int colBase, int swave,
                                                 int kb, int buf) {
#if USE_TDM
    // 2-D tensor descriptor (D#), ISA 8.3/8.4: dim0 = K (contiguous), dim1 = N rows
    unsigned long long ga = (unsigned long long)(size_t)Bt +
        ((unsigned long long)(unsigned)(colBase + swave * 8) * (unsigned)ldb +
         (unsigned)kb) * 2ull;
    unsigned lds = (unsigned)(buf * BTILE_ELEMS * 2 + swave * 8 * 64);
    v4u g0;
    g0.x = 1u;                                         // count=1, user mode
    g0.y = lds;                                        // lds_addr (bytes)
    g0.z = (unsigned)ga;                               // global_addr[31:0]
    g0.w = ((unsigned)(ga >> 32) & 0x01FFFFFFu) | 0x80000000u; // ga[56:32] | type=2
    v8i g1;
    g1[0] = 0x00010000;                                // wg_mask=0, data_size=1 (2B)
    g1[1] = (int)(((unsigned)ldb & 0xFFFFu) << 16);    // tensor_dim0[15:0] @ bit48
    g1[2] = (int)((((unsigned)ldb >> 16) & 0xFFFFu) |  // tensor_dim0[31:16]
                  (((unsigned)Brows & 0xFFFFu) << 16));// tensor_dim1[15:0]
    g1[3] = (int)((((unsigned)Brows >> 16) & 0xFFFFu) |
                  (32u << 16));                        // tile_dim0 = 32 (K)
    g1[4] = 8;                                         // tile_dim1 = 8 cols, tile_dim2=0
    g1[5] = ldb;                                       // tensor_dim0_stride[31:0]
    g1[6] = 0;                                         // stride hi, dim1_stride lo
    g1[7] = 0;
    v4i z4 = {0, 0, 0, 0};
    v8i z8 = {0, 0, 0, 0, 0, 0, 0, 0};
    __builtin_amdgcn_tensor_load_to_lds(g0, g1, z4, z4, z8, 0);
#endif
}

__global__ void __launch_bounds__(256)
gemm_bf16_kernel(const unsigned short* __restrict__ A, int lda,
                 const unsigned short* __restrict__ Bt, int ldb, int Brows,
                 float* __restrict__ C, int ldc, int K) {
    __shared__ unsigned short btile[2][BTILE_ELEMS];

    const int tid  = threadIdx.x;
    const int wave = tid >> 5;
    const int lane = tid & 31;
    const int rowBase = blockIdx.y * 128 + wave * 16;
    const int colBase = blockIdx.x * 64;

    const int mrow = rowBase + (lane & 15);
    const int klo  = (lane >> 4) ? 8 : 0;
    const int c0   = lane & 15;

    v8f acc0 = {}, acc1 = {}, acc2 = {}, acc3 = {};
    const unsigned short* arow = A + (size_t)mrow * lda;

    // Zero-init both LDS buffers so the compiler sees btile as written: the
    // TDM engine's stores are invisible to alias analysis, which otherwise
    // folds the ds_load_b128 fragment reads to undef and deletes them.
    {
        unsigned short* bt0 = &btile[0][0];
        float4 z = {};
#pragma unroll
        for (int i = 0; i < 4; ++i)
            *reinterpret_cast<float4*>(bt0 + (tid * 4 + i) * 8) = z;
    }
    __syncthreads();

#if USE_TDM
    const int swave = __builtin_amdgcn_readfirstlane(wave);
    tdm_issue_bslice(Bt, ldb, Brows, colBase, swave, 0, 0);
#endif

    int cur = 0;
    for (int kb = 0; kb < K; kb += 32) {
        const bool hasNext = (kb + 32) < K;
#if USE_TDM
        if (hasNext) {
            tdm_issue_bslice(Bt, ldb, Brows, colBase, swave, kb + 32, cur ^ 1);
            __builtin_amdgcn_s_wait_tensorcnt(1);   // current tile complete
        } else {
            __builtin_amdgcn_s_wait_tensorcnt(0);
        }
#else
        {   // cooperative fallback: 256 threads x 16B cover the 4KB tile
            int n = tid >> 2, seg = tid & 3;
            *reinterpret_cast<float4*>(&btile[cur][n * 32 + seg * 8]) =
                *reinterpret_cast<const float4*>(Bt + (size_t)(colBase + n) * ldb + kb + seg * 8);
        }
#endif
        __syncthreads();

        const int k0 = kb + klo;
        FragBF a, f0, f1, f2, f3;
        a.f4[0] = *reinterpret_cast<const float4*>(arow + k0);
        a.f4[1] = *reinterpret_cast<const float4*>(arow + k0 + 16);
        if (hasNext)  // lowers to global_prefetch_b8
            __builtin_prefetch(arow + k0 + 32, 0, 1);

        const unsigned short* bt = btile[cur];
        f0.f4[0] = *reinterpret_cast<const float4*>(&bt[(c0 +  0) * 32 + klo]);
        f0.f4[1] = *reinterpret_cast<const float4*>(&bt[(c0 +  0) * 32 + klo + 16]);
        f1.f4[0] = *reinterpret_cast<const float4*>(&bt[(c0 + 16) * 32 + klo]);
        f1.f4[1] = *reinterpret_cast<const float4*>(&bt[(c0 + 16) * 32 + klo + 16]);
        f2.f4[0] = *reinterpret_cast<const float4*>(&bt[(c0 + 32) * 32 + klo]);
        f2.f4[1] = *reinterpret_cast<const float4*>(&bt[(c0 + 32) * 32 + klo + 16]);
        f3.f4[0] = *reinterpret_cast<const float4*>(&bt[(c0 + 48) * 32 + klo]);
        f3.f4[1] = *reinterpret_cast<const float4*>(&bt[(c0 + 48) * 32 + klo + 16]);

        acc0 = __builtin_amdgcn_wmma_f32_16x16x32_bf16(false, a.v, false, f0.v,
                                                       (short)0, acc0, false, false);
        acc1 = __builtin_amdgcn_wmma_f32_16x16x32_bf16(false, a.v, false, f1.v,
                                                       (short)0, acc1, false, false);
        acc2 = __builtin_amdgcn_wmma_f32_16x16x32_bf16(false, a.v, false, f2.v,
                                                       (short)0, acc2, false, false);
        acc3 = __builtin_amdgcn_wmma_f32_16x16x32_bf16(false, a.v, false, f3.v,
                                                       (short)0, acc3, false, false);

        __syncthreads();   // all waves done reading btile[cur] before it is re-DMAed
        cur ^= 1;
    }

    // C/D layout: VGPR r -> (M = r + 8*(lane>=16), N = lane&15)
    const int crow = rowBase + ((lane >> 4) ? 8 : 0);
    const int ccol = colBase + c0;
#pragma unroll
    for (int r = 0; r < 8; ++r) {
        C[(size_t)(crow + r) * ldc + ccol +  0] = acc0[r];
        C[(size_t)(crow + r) * ldc + ccol + 16] = acc1[r];
        C[(size_t)(crow + r) * ldc + ccol + 32] = acc2[r];
        C[(size_t)(crow + r) * ldc + ccol + 48] = acc3[r];
    }
}

// ---------------------------------------------------------------------------
// tanh(fp32) -> bf16
// ---------------------------------------------------------------------------
__global__ void tanh_bf16_kernel(const float* __restrict__ in,
                                 unsigned short* __restrict__ out, int n) {
    int idx = blockIdx.x * blockDim.x + threadIdx.x;
    if (idx < n) out[idx] = f2bf(tanhf(in[idx]));
}

// ---------------------------------------------------------------------------
// x{w,k,v,r} = bf16(x + xx*(maa_* + m_*))
// ---------------------------------------------------------------------------
__global__ void mix_kernel(const float* __restrict__ x,  const float* __restrict__ xx,
                           const float* __restrict__ aw, const float* __restrict__ ak,
                           const float* __restrict__ av, const float* __restrict__ ar,
                           const float* __restrict__ mw, const float* __restrict__ mk,
                           const float* __restrict__ mv, const float* __restrict__ mr,
                           unsigned short* __restrict__ xw, unsigned short* __restrict__ xk,
                           unsigned short* __restrict__ xv, unsigned short* __restrict__ xr) {
    int idx = blockIdx.x * blockDim.x + threadIdx.x;
    if (idx >= M_ * C_) return;
    int c = idx % C_;
    float xb = x[idx], xd = xx[idx];
    xw[idx] = f2bf(xb + xd * (aw[c] + mw[idx]));
    xk[idx] = f2bf(xb + xd * (ak[c] + mk[idx]));
    xv[idx] = f2bf(xb + xd * (av[c] + mv[idx]));
    xr[idx] = f2bf(xb + xd * (ar[c] + mr[idx]));
}

// ---------------------------------------------------------------------------
// w = exp(-exp(time_decay[c] + ww[idx]))
// ---------------------------------------------------------------------------
__global__ void decay_kernel(const float* __restrict__ ww,
                             const float* __restrict__ td,
                             float* __restrict__ w) {
    int idx = blockIdx.x * blockDim.x + threadIdx.x;
    if (idx >= M_ * C_) return;
    w[idx] = __expf(-__expf(td[idx % C_] + ww[idx]));
}

// ---------------------------------------------------------------------------
// WKV6 recurrence: one block per (b,h), 64 threads; thread j owns state
// column S[i][j] (64 VGPRs). r/k/w broadcast per step through LDS.
// ---------------------------------------------------------------------------
__global__ void __launch_bounds__(64)
wkv6_kernel(const float* __restrict__ r, const float* __restrict__ k,
            const float* __restrict__ v, const float* __restrict__ w,
            const float* __restrict__ u, float* __restrict__ y) {
    const int bh = blockIdx.x;
    const int b  = bh / H_;
    const int h  = bh % H_;
    const int j  = threadIdx.x;

    __shared__ float sr[HEAD_], sk[HEAD_], sw[HEAD_], su[HEAD_];
    float S[HEAD_];
#pragma unroll
    for (int i = 0; i < HEAD_; ++i) S[i] = 0.f;
    su[j] = u[h * HEAD_ + j];
    __syncthreads();

    const size_t base = (size_t)(b * T_) * C_ + h * HEAD_;
    for (int t = 0; t < T_; ++t) {
        const size_t off = base + (size_t)t * C_;
        sr[j] = r[off + j];
        sk[j] = k[off + j];
        sw[j] = w[off + j];
        const float vj = v[off + j];
        __syncthreads();
        float acc = 0.f;
#pragma unroll 8
        for (int i = 0; i < HEAD_; ++i) {
            float kv = sk[i] * vj;
            acc += sr[i] * (S[i] + su[i] * kv);
            S[i] = sw[i] * S[i] + kv;
        }
        y[off + j] = acc;
        __syncthreads();
    }
}

// ---------------------------------------------------------------------------
// GroupNorm over each head of 64 + affine; output bf16 for final GEMM
// ---------------------------------------------------------------------------
__global__ void __launch_bounds__(64)
gnorm_kernel(const float* __restrict__ y, const float* __restrict__ lw,
             const float* __restrict__ lb, unsigned short* __restrict__ yn) {
    const int rh = blockIdx.x;          // m*H + h
    const int m  = rh / H_;
    const int h  = rh % H_;
    const int j  = threadIdx.x;
    __shared__ float red[HEAD_];

    const size_t idx = (size_t)m * C_ + h * HEAD_ + j;
    const float val = y[idx];
    red[j] = val;
    __syncthreads();
    for (int s = 32; s > 0; s >>= 1) { if (j < s) red[j] += red[j + s]; __syncthreads(); }
    const float mean = red[0] * (1.f / HEAD_);
    __syncthreads();
    const float d = val - mean;
    red[j] = d * d;
    __syncthreads();
    for (int s = 32; s > 0; s >>= 1) { if (j < s) red[j] += red[j + s]; __syncthreads(); }
    const float var = red[0] * (1.f / HEAD_);
    const float rs  = rsqrtf(var + 64e-5f);
    const int c = h * HEAD_ + j;
    yn[idx] = f2bf(d * rs * lw[c] + lb[c]);
}

// ---------------------------------------------------------------------------
extern "C" void kernel_launch(void* const* d_in, const int* in_sizes, int n_in,
                              void* d_out, int out_size, void* d_ws, size_t ws_size,
                              hipStream_t stream) {
    const float* x       = (const float*)d_in[0];   // query
    const float* maa_x   = (const float*)d_in[3];
    const float* maa_w   = (const float*)d_in[4];
    const float* maa_k   = (const float*)d_in[5];
    const float* maa_v   = (const float*)d_in[6];
    const float* maa_r   = (const float*)d_in[7];
    const float* maa_w1  = (const float*)d_in[8];   // [C, 128]
    const float* maa_w2  = (const float*)d_in[9];   // [4, 32, C]
    const float* tdecay  = (const float*)d_in[10];
    const float* dec_w1  = (const float*)d_in[11];  // [C, 64]
    const float* dec_w2  = (const float*)d_in[12];  // [64, C]
    const float* faaaa   = (const float*)d_in[13];  // [H, 64]
    const float* W_r     = (const float*)d_in[14];
    const float* W_k     = (const float*)d_in[15];
    const float* W_v     = (const float*)d_in[16];
    const float* W_o     = (const float*)d_in[17];
    const float* ln_w    = (const float*)d_in[18];
    const float* ln_b    = (const float*)d_in[19];
    float* out = (float*)d_out;

    // ---- workspace carve-out (256B aligned) ----
    char* wsb = (char*)d_ws;
    size_t off = 0;
    auto carve = [&](size_t bytes) -> char* {
        off = (off + 255) & ~(size_t)255;
        char* p = wsb + off;
        off += bytes;
        return p;
    };
    const size_t MC = (size_t)M_ * C_;
    float*          xx      = (float*)         carve(MC * 4);
    unsigned short* xxx     = (unsigned short*)carve(MC * 2);
    unsigned short* wrt     = (unsigned short*)carve((size_t)C_ * C_ * 2);
    unsigned short* wkt     = (unsigned short*)carve((size_t)C_ * C_ * 2);
    unsigned short* wvt     = (unsigned short*)carve((size_t)C_ * C_ * 2);
    unsigned short* wot     = (unsigned short*)carve((size_t)C_ * C_ * 2);
    unsigned short* w1t     = (unsigned short*)carve((size_t)C_ * 128 * 2);   // [128, C]
    unsigned short* w2t     = (unsigned short*)carve((size_t)4 * C_ * 32 * 2);// 4 x [C, 32]
    unsigned short* dw1t    = (unsigned short*)carve((size_t)C_ * 64 * 2);    // [64, C]
    unsigned short* dw2t    = (unsigned short*)carve((size_t)C_ * 64 * 2);    // [C, 64]
    float*          mpre    = (float*)         carve((size_t)M_ * 128 * 4);
    unsigned short* mbf     = (unsigned short*)carve((size_t)M_ * 128 * 2);
    float*          mixW    = (float*)         carve(MC * 4);
    float*          mixK    = (float*)         carve(MC * 4);
    float*          mixV    = (float*)         carve(MC * 4);
    float*          mixR    = (float*)         carve(MC * 4);
    unsigned short* xw      = (unsigned short*)carve(MC * 2);
    unsigned short* xk      = (unsigned short*)carve(MC * 2);
    unsigned short* xv      = (unsigned short*)carve(MC * 2);
    unsigned short* xr      = (unsigned short*)carve(MC * 2);
    float*          rbuf    = (float*)         carve(MC * 4);
    float*          kbuf    = (float*)         carve(MC * 4);
    float*          vbuf    = (float*)         carve(MC * 4);
    float*          t1pre   = (float*)         carve((size_t)M_ * 64 * 4);
    unsigned short* t1bf    = (unsigned short*)carve((size_t)M_ * 64 * 2);
    float*          wwbuf   = (float*)         carve(MC * 4);
    float*          wdec    = (float*)         carve(MC * 4);
    float*          ybuf    = (float*)         carve(MC * 4);
    unsigned short* ynbf    = (unsigned short*)carve(MC * 2);
    (void)ws_size; (void)n_in; (void)in_sizes; (void)out_size;

    const int EW = 256;
    const int nMC = (int)MC;

    // 1) weight convert + transpose -> bf16 N-major
    convT_kernel<<<(C_*C_ + EW-1)/EW, EW, 0, stream>>>(W_r, wrt, C_, C_);
    convT_kernel<<<(C_*C_ + EW-1)/EW, EW, 0, stream>>>(W_k, wkt, C_, C_);
    convT_kernel<<<(C_*C_ + EW-1)/EW, EW, 0, stream>>>(W_v, wvt, C_, C_);
    convT_kernel<<<(C_*C_ + EW-1)/EW, EW, 0, stream>>>(W_o, wot, C_, C_);
    convT_kernel<<<(C_*128 + EW-1)/EW, EW, 0, stream>>>(maa_w1, w1t, C_, 128);
    for (int f = 0; f < 4; ++f)
        convT_kernel<<<(32*C_ + EW-1)/EW, EW, 0, stream>>>(
            maa_w2 + (size_t)f * 32 * C_, w2t + (size_t)f * C_ * 32, 32, C_);
    convT_kernel<<<(C_*64 + EW-1)/EW, EW, 0, stream>>>(dec_w1, dw1t, C_, 64);
    convT_kernel<<<(64*C_ + EW-1)/EW, EW, 0, stream>>>(dec_w2, dw2t, 64, C_);

    // 2) token shift
    prep_kernel<<<(nMC + EW-1)/EW, EW, 0, stream>>>(x, maa_x, xx, xxx);

    // 3) LoRA mix: mpre = xxx @ maa_w1 ; m = tanh(mpre)
    gemm_bf16_kernel<<<dim3(128/64, M_/128), 256, 0, stream>>>(
        xxx, C_, w1t, C_, 128, mpre, 128, C_);
    tanh_bf16_kernel<<<(M_*128 + EW-1)/EW, EW, 0, stream>>>(mpre, mbf, M_*128);

    // 4) mix_f = m_f @ w2_f  (K=32)
    float* mixes[4] = {mixW, mixK, mixV, mixR};
    for (int f = 0; f < 4; ++f)
        gemm_bf16_kernel<<<dim3(C_/64, M_/128), 256, 0, stream>>>(
            mbf + f*32, 128, w2t + (size_t)f * C_ * 32, 32, C_, mixes[f], C_, 32);

    // 5) x{w,k,v,r}
    mix_kernel<<<(nMC + EW-1)/EW, EW, 0, stream>>>(
        x, xx, maa_w, maa_k, maa_v, maa_r, mixW, mixK, mixV, mixR, xw, xk, xv, xr);

    // 6) projections r/k/v
    gemm_bf16_kernel<<<dim3(C_/64, M_/128), 256, 0, stream>>>(xr, C_, wrt, C_, C_, rbuf, C_, C_);
    gemm_bf16_kernel<<<dim3(C_/64, M_/128), 256, 0, stream>>>(xk, C_, wkt, C_, C_, kbuf, C_, C_);
    gemm_bf16_kernel<<<dim3(C_/64, M_/128), 256, 0, stream>>>(xv, C_, wvt, C_, C_, vbuf, C_, C_);

    // 7) decay path: t1 = tanh(xw @ dec_w1); ww = t1 @ dec_w2; w = exp(-exp(td+ww))
    gemm_bf16_kernel<<<dim3(64/64, M_/128), 256, 0, stream>>>(xw, C_, dw1t, C_, 64, t1pre, 64, C_);
    tanh_bf16_kernel<<<(M_*64 + EW-1)/EW, EW, 0, stream>>>(t1pre, t1bf, M_*64);
    gemm_bf16_kernel<<<dim3(C_/64, M_/128), 256, 0, stream>>>(t1bf, 64, dw2t, 64, C_, wwbuf, C_, 64);
    decay_kernel<<<(nMC + EW-1)/EW, EW, 0, stream>>>(wwbuf, tdecay, wdec);

    // 8) WKV recurrence
    wkv6_kernel<<<B_ * H_, 64, 0, stream>>>(rbuf, kbuf, vbuf, wdec, faaaa, ybuf);

    // 9) GroupNorm -> bf16
    gnorm_kernel<<<M_ * H_, 64, 0, stream>>>(ybuf, ln_w, ln_b, ynbf);

    // 10) output projection -> d_out (fp32)
    gemm_bf16_kernel<<<dim3(C_/64, M_/128), 256, 0, stream>>>(ynbf, C_, wot, C_, C_, out, C_, C_);
}
